// VoxelJAFAR_72060961292755
// MI455X (gfx1250) — matrix operand order repaired and checked by reading, not compiled, not run on hardware
//
#include <hip/hip_runtime.h>
#include <hip/hip_bf16.h>

// ---------------- types ----------------
typedef __attribute__((ext_vector_type(16))) __bf16 v16bf;
typedef __attribute__((ext_vector_type(8)))  __bf16 v8bf;
typedef __attribute__((ext_vector_type(8)))  float  v8f;
typedef __attribute__((ext_vector_type(4)))  float  v4f;

#define M_TOT 8192
#define GEOC  32
#define SEMC  96
#define DD    64
#define KS3   125
#define R3    343
#define R3P   352     // padded to 22*16
#define NC    13
#define GRD   32
#define NCELL (2*32*32*32)   // 65536 dense cells

// native converts: clang legalizes fptrunc f32->bf16 to the hw cvt op on gfx1250
static __device__ __forceinline__ __bf16 f2bf(float f) { return (__bf16)f; }
static __device__ __forceinline__ float  bf2f(__bf16 h) { return (float)h; }
static __device__ __forceinline__ __bf16 bfz() {
  unsigned short z = 0;
  return __builtin_bit_cast(__bf16, z);
}
static __device__ __forceinline__ v16bf mkv16(v8bf lo, v8bf hi) {
  return __builtin_shufflevector(lo, hi, 0,1,2,3,4,5,6,7,8,9,10,11,12,13,14,15);
}
#define WMMA_BF16(A,B,C) __builtin_amdgcn_wmma_f32_16x16x32_bf16(false,(A),false,(B),(short)0,(C),false,false)

// ---------------- hash / LUT kernels ----------------
__global__ void k_init_lut(int* __restrict__ lut) {
  int i = blockIdx.x * blockDim.x + threadIdx.x;
  if (i < NCELL) lut[i] = -1;
}

__global__ void k_scatter_lut(const int* __restrict__ idx, int* __restrict__ lut) {
  int i = blockIdx.x * blockDim.x + threadIdx.x;
  if (i < M_TOT) {
    int b = idx[i*4+0], x = idx[i*4+1], y = idx[i*4+2], z = idx[i*4+3];
    lut[((b*GRD + z)*GRD + y)*GRD + x] = i;
  }
}

// exclusive prefix of occupancy over the 65536-cell grid (single block)
__global__ __launch_bounds__(1024) void k_scan(const int* __restrict__ lut, int* __restrict__ rank) {
  __shared__ int part[1024];
  int tid  = threadIdx.x;
  int base = tid * 64;
  int s = 0;
  for (int j = 0; j < 64; ++j) s += (lut[base + j] >= 0) ? 1 : 0;
  part[tid] = s;
  __syncthreads();
  for (int off = 1; off < 1024; off <<= 1) {
    int v = (tid >= off) ? part[tid - off] : 0;
    __syncthreads();
    part[tid] += v;
    __syncthreads();
  }
  int run = part[tid] - s;   // exclusive prefix
  for (int j = 0; j < 64; ++j) {
    rank[base + j] = run;
    run += (lut[base + j] >= 0) ? 1 : 0;
  }
}

__global__ void k_sorted(const int* __restrict__ idx, const int* __restrict__ rank,
                         long long* __restrict__ sk, int* __restrict__ si) {
  int i = blockIdx.x * blockDim.x + threadIdx.x;
  if (i < M_TOT) {
    int b = idx[i*4+0], x = idx[i*4+1], y = idx[i*4+2], z = idx[i*4+3];
    int d = ((b*GRD + z)*GRD + y)*GRD + x;
    int r = rank[d];
    long long key = (long long)b*68719476736LL + (long long)z*16777216LL
                  + (long long)y*4096LL + (long long)x;
    sk[r] = key;
    si[r] = i;
  }
}

// ---------------- one-shot weight pre-conversion to bf16 B-layout ([n][k]) ----
__global__ void k_wprep(const float* __restrict__ Wconv, const float* __restrict__ Wq,
                        const float* __restrict__ Wk, const float* __restrict__ Wv,
                        const float* __restrict__ Wout, const float* __restrict__ Wcls,
                        const float* __restrict__ pemb,
                        __bf16* __restrict__ Wc, __bf16* __restrict__ Bq,
                        __bf16* __restrict__ Bk, __bf16* __restrict__ Bv,
                        __bf16* __restrict__ Bo, __bf16* __restrict__ Bc,
                        __bf16* __restrict__ Pb) {
  int i = blockIdx.x * blockDim.x + threadIdx.x;
  if (i < KS3 * GEOC * DD) {            // [t][k][n] -> [t][n][k]
    int t = i >> 11, r = i & 2047;
    int k = r >> 6, n = r & 63;
    Wc[t * 2048 + n * GEOC + k] = f2bf(Wconv[i]);
  }
  if (i < DD * DD) {                    // [k][n] -> [n][k]
    int k = i >> 6, n = i & 63;
    Bq[n * DD + k] = f2bf(Wq[i]);
    Bk[n * DD + k] = f2bf(Wk[i]);
    Bo[n * DD + k] = f2bf(Wout[i]);
  }
  if (i < SEMC * DD) {
    int k = i >> 6, n = i & 63;
    Bv[n * SEMC + k] = f2bf(Wv[i]);
  }
  if (i < DD * 16) {                    // W_cls padded to 16 cols
    int k = i >> 4, n = i & 15;
    Bc[n * DD + k] = (n < NC) ? f2bf(Wcls[k * NC + n]) : bfz();
  }
  if (i < R3 * DD) {                    // pos_emb already [n][k]
    Pb[i] = f2bf(pemb[i]);
  }
}

// ---------------- conv (hash 5^3) + LN + ReLU + boundary head ----------------
// per wave: 16 rows x 64 cols, one v_wmma_f32_16x16x32_bf16 per tap per n-tile
__global__ __launch_bounds__(128) void k_conv(
    const int* __restrict__ idx, const float* __restrict__ geo,
    const __bf16* __restrict__ Wc, const float* __restrict__ gamma,
    const float* __restrict__ beta, const float* __restrict__ Wbdy,
    const float* __restrict__ bbdy, const int* __restrict__ lut,
    __bf16* __restrict__ qgeo_bf, float* __restrict__ out_bdy) {
  __shared__ float Cs[4][16 * DD];     // per-wave C spill
  int tid   = threadIdx.x;
  int wave  = tid >> 5;
  int lane  = tid & 31;
  int row   = lane & 15;
  int khalf = lane >> 4;
  int m0    = (blockIdx.x * 4 + wave) * 16;
  int m     = m0 + row;
  int cb = idx[m*4+0], cx = idx[m*4+1], cy = idx[m*4+2], cz = idx[m*4+3];

  v8f acc[4] = {};

  const __bf16* Wt = Wc;
  for (int dx = -2; dx <= 2; ++dx)
    for (int dy = -2; dy <= 2; ++dy)
      for (int dz = -2; dz <= 2; ++dz, Wt += 2048) {
        __builtin_prefetch(Wt + 2048 + lane * 64, 0, 1);

        int nx = cx + dx, ny = cy + dy, nz = cz + dz;
        int nidx = -1;
        if ((unsigned)nx < 32u && (unsigned)ny < 32u && (unsigned)nz < 32u)
          nidx = lut[((cb*GRD + nz)*GRD + ny)*GRD + nx];
        // wave-uniform skip when the whole tile has no occupied neighbor here
        if (__ballot(nidx >= 0) == 0ull) continue;

        // gather A: row = lane&15, K chunks {kb..kb+7, 16+kb..16+kb+7}, kb=8*khalf
        v16bf a;
        if (nidx >= 0) {
          const float* gr = geo + (size_t)nidx * GEOC + khalf * 8;
          v4f g0 = *(const v4f*)(gr);
          v4f g1 = *(const v4f*)(gr + 4);
          v4f g2 = *(const v4f*)(gr + 16);
          v4f g3 = *(const v4f*)(gr + 20);
          #pragma unroll
          for (int j = 0; j < 4; ++j) {
            a[j] = f2bf(g0[j]); a[4+j] = f2bf(g1[j]);
            a[8+j] = f2bf(g2[j]); a[12+j] = f2bf(g3[j]);
          }
        } else {
          #pragma unroll
          for (int j = 0; j < 16; ++j) a[j] = bfz();
        }

        // preload all 4 B tiles, then issue the 4 WMMAs
        v16bf bt[4];
        #pragma unroll
        for (int nt = 0; nt < 4; ++nt) {
          const __bf16* bp = Wt + (nt * 16 + (lane & 15)) * GEOC + khalf * 8;
          bt[nt] = mkv16(*(const v8bf*)bp, *(const v8bf*)(bp + 16));
        }
        #pragma unroll
        for (int nt = 0; nt < 4; ++nt)
          acc[nt] = WMMA_BF16(a, bt[nt], acc[nt]);
      }

  // spill C -> LDS (C layout: elem i => row i+8*khalf, col nt*16+(lane&15))
  float* cs = &Cs[wave][0];
  #pragma unroll
  for (int nt = 0; nt < 4; ++nt)
    #pragma unroll
    for (int i = 0; i < 8; ++i)
      cs[(i + 8 * khalf) * DD + nt * 16 + (lane & 15)] = acc[nt][i];
  __syncthreads();

  // LayerNorm + ReLU + boundary head: one lane per row
  if (lane < 16) {
    const float* r = cs + lane * DD;
    float mu = 0.f;
    for (int d = 0; d < DD; ++d) mu += r[d];
    mu *= (1.0f / DD);
    float var = 0.f;
    for (int d = 0; d < DD; ++d) { float tdv = r[d] - mu; var += tdv * tdv; }
    var *= (1.0f / DD);
    float inv = rsqrtf(var + 1e-5f);
    float bd = 0.f;
    int mo = m0 + lane;
    for (int d = 0; d < DD; ++d) {
      float v = (r[d] - mu) * inv * gamma[d] + beta[d];
      v = fmaxf(v, 0.0f);
      qgeo_bf[(size_t)mo * DD + d] = f2bf(v);
      bd += v * Wbdy[d];
    }
    out_bdy[mo] = bd + bbdy[0];
  }
}

// ---------------- q/k/v projections ----------------
__global__ __launch_bounds__(128) void k_qkv(
    const __bf16* __restrict__ qgeo_bf, const float* __restrict__ sem,
    const __bf16* __restrict__ Bq, const __bf16* __restrict__ Bk,
    const __bf16* __restrict__ Bv,
    __bf16* __restrict__ qb, float* __restrict__ qf,
    float* __restrict__ kf, float* __restrict__ vf) {
  int tid   = threadIdx.x;
  int wave  = tid >> 5;
  int lane  = tid & 31;
  int row   = lane & 15;
  int khalf = lane >> 4;
  int m0    = (blockIdx.x * 4 + wave) * 16;

  v8f aq[4] = {}, ak[4] = {}, av[4] = {};

  // q,k : K = 64 (2 chunks of 32)
  #pragma unroll
  for (int c = 0; c < 2; ++c) {
    const __bf16* ar = qgeo_bf + (size_t)(m0 + row) * DD + c * 32 + khalf * 8;
    v16bf a = mkv16(*(const v8bf*)ar, *(const v8bf*)(ar + 16));
    v16bf bq[4], bk[4];
    #pragma unroll
    for (int nt = 0; nt < 4; ++nt) {
      const __bf16* bp  = Bq + (nt * 16 + (lane & 15)) * DD + c * 32 + khalf * 8;
      const __bf16* bp2 = Bk + (nt * 16 + (lane & 15)) * DD + c * 32 + khalf * 8;
      bq[nt] = mkv16(*(const v8bf*)bp,  *(const v8bf*)(bp + 16));
      bk[nt] = mkv16(*(const v8bf*)bp2, *(const v8bf*)(bp2 + 16));
    }
    #pragma unroll
    for (int nt = 0; nt < 4; ++nt) {
      aq[nt] = WMMA_BF16(a, bq[nt], aq[nt]);
      ak[nt] = WMMA_BF16(a, bk[nt], ak[nt]);
    }
  }
  // v : K = 96 (3 chunks of 32), A from sem f32
  #pragma unroll
  for (int c = 0; c < 3; ++c) {
    const float* sr = sem + (size_t)(m0 + row) * SEMC + c * 32 + khalf * 8;
    v4f g0 = *(const v4f*)(sr);
    v4f g1 = *(const v4f*)(sr + 4);
    v4f g2 = *(const v4f*)(sr + 16);
    v4f g3 = *(const v4f*)(sr + 20);
    v16bf a;
    #pragma unroll
    for (int j = 0; j < 4; ++j) {
      a[j] = f2bf(g0[j]); a[4+j] = f2bf(g1[j]);
      a[8+j] = f2bf(g2[j]); a[12+j] = f2bf(g3[j]);
    }
    v16bf bv[4];
    #pragma unroll
    for (int nt = 0; nt < 4; ++nt) {
      const __bf16* bp = Bv + (nt * 16 + (lane & 15)) * SEMC + c * 32 + khalf * 8;
      bv[nt] = mkv16(*(const v8bf*)bp, *(const v8bf*)(bp + 16));
    }
    #pragma unroll
    for (int nt = 0; nt < 4; ++nt)
      av[nt] = WMMA_BF16(a, bv[nt], av[nt]);
  }
  // store q bf16+f32, k f32, v f32
  #pragma unroll
  for (int nt = 0; nt < 4; ++nt)
    #pragma unroll
    for (int i = 0; i < 8; ++i) {
      int rr = i + 8 * khalf, cc = nt * 16 + (lane & 15);
      size_t o = (size_t)(m0 + rr) * DD + cc;
      qb[o] = f2bf(aq[nt][i]);
      qf[o] = aq[nt][i];
      kf[o] = ak[nt][i];
      vf[o] = av[nt][i];
    }
}

// ---------------- attention (one wave per 16-row tile) ----------------
__global__ __launch_bounds__(32) void k_attn(
    const int* __restrict__ idx, const int* __restrict__ lut,
    const int* __restrict__ rank,
    const long long* __restrict__ sk, const int* __restrict__ si,
    const __bf16* __restrict__ qb, const float* __restrict__ qf,
    const float* __restrict__ kf, const float* __restrict__ vf,
    const __bf16* __restrict__ Pb,
    float* __restrict__ out_aff, int* __restrict__ out_nbr,
    float* __restrict__ refined) {
  __shared__ float lg[16 * R3P];   // pos-bias -> logits -> aff
  __shared__ int   nb[16 * R3P];
  int lane  = threadIdx.x;
  int row   = lane & 15;
  int khalf = lane >> 4;
  int m0    = blockIdx.x * 16;

  // ---- pos bias via WMMA: q_tile(16x64) x pos_emb^T(64x343) ----
  v16bf aq2[2];
  #pragma unroll
  for (int c = 0; c < 2; ++c) {
    const __bf16* ar = qb + (size_t)(m0 + row) * DD + c * 32 + khalf * 8;
    aq2[c] = mkv16(*(const v8bf*)ar, *(const v8bf*)(ar + 16));
  }
  #pragma unroll 1
  for (int nt = 0; nt < 22; ++nt) {
    v8f acc = {};
    int n = nt * 16 + (lane & 15);
    v16bf b0, b1;
    if (n < R3) {
      const __bf16* bp0 = Pb + (size_t)n * DD + khalf * 8;
      const __bf16* bp1 = bp0 + 32;
      b0 = mkv16(*(const v8bf*)bp0, *(const v8bf*)(bp0 + 16));
      b1 = mkv16(*(const v8bf*)bp1, *(const v8bf*)(bp1 + 16));
    } else {
      #pragma unroll
      for (int j = 0; j < 16; ++j) { b0[j] = bfz(); b1[j] = bfz(); }
    }
    acc = WMMA_BF16(aq2[0], b0, acc);
    acc = WMMA_BF16(aq2[1], b1, acc);
    #pragma unroll
    for (int i = 0; i < 8; ++i)
      lg[(i + 8 * khalf) * R3P + n] = acc[i];
  }
  __syncthreads();

  // ---- per-row gathered attention ----
  for (int rr = 0; rr < 16; ++rr) {
    int mI = m0 + rr;
    // q row in registers (f32)
    float qv[DD];
    const float* qr = qf + (size_t)mI * DD;
    #pragma unroll
    for (int j = 0; j < 16; ++j) {
      v4f q4 = *(const v4f*)(qr + j * 4);
      #pragma unroll
      for (int e = 0; e < 4; ++e) qv[j * 4 + e] = q4[e];
    }
    int b_ = idx[mI*4+0], x_ = idx[mI*4+1], y_ = idx[mI*4+2], z_ = idx[mI*4+3];

    float mx = -3.0e38f;
    for (int t = lane; t < R3P; t += 32) {
      float lgt = -1.0e9f;
      int nout = 0;
      if (t < R3) {
        int dx = t / 49 - 3, dy = (t / 7) % 7 - 3, dz = t % 7 - 3;
        int nx = x_ + dx, ny = y_ + dy, nz = z_ + dz;
        bool inb = (unsigned)nx < 32u && (unsigned)ny < 32u && (unsigned)nz < 32u;
        int dense = ((b_*GRD + nz)*GRD + ny)*GRD + nx;
        int nidx = inb ? lut[dense] : -1;
        if (nidx >= 0) {
          const float* kr = kf + (size_t)nidx * DD;
          float dot = 0.f;
          #pragma unroll
          for (int j = 0; j < 16; ++j) {
            v4f k4 = *(const v4f*)(kr + j * 4);
            dot += qv[j*4+0]*k4[0] + qv[j*4+1]*k4[1]
                 + qv[j*4+2]*k4[2] + qv[j*4+3]*k4[3];
          }
          lgt = (dot + lg[rr * R3P + t]) * 0.125f;   // 1/sqrt(64)
          nout = nidx | 0x40000000;
        } else {
          // searchsorted replication for masked neighbors (nbr output only)
          int pos;
          if (inb) {
            // empty in-range cell: key order == dense order, so
            // searchsorted == #occupied cells below == rank[dense]
            pos = rank[dense];
          } else {
            long long qk = (long long)b_*68719476736LL + (long long)nz*16777216LL
                         + (long long)ny*4096LL + (long long)nx;
            int lo = 0, hi = M_TOT;
            while (lo < hi) { int mid = (lo + hi) >> 1; if (sk[mid] < qk) lo = mid + 1; else hi = mid; }
            pos = lo;
          }
          if (pos > M_TOT - 1) pos = M_TOT - 1;
          nout = si[pos];
        }
      }
      lg[rr * R3P + t] = lgt;
      nb[rr * R3P + t] = nout;
      mx = fmaxf(mx, lgt);
    }
    #pragma unroll
    for (int off = 16; off > 0; off >>= 1) mx = fmaxf(mx, __shfl_xor(mx, off, 32));

    float sum = 0.f;
    for (int t = lane; t < R3; t += 32) {
      float e = __expf(lg[rr * R3P + t] - mx);
      lg[rr * R3P + t] = e;
      sum += e;
    }
    #pragma unroll
    for (int off = 16; off > 0; off >>= 1) sum += __shfl_xor(sum, off, 32);
    float inv = 1.0f / sum;

    for (int t = lane; t < R3; t += 32) {
      float a = lg[rr * R3P + t] * inv;
      lg[rr * R3P + t] = a;
      out_aff[(size_t)mI * R3 + t] = a;
      out_nbr[(size_t)mI * R3 + t] = nb[rr * R3P + t] & 0x3fffffff;
    }
    __syncthreads();

    // refined = sum_j aff_j * v[nbr_j] + v_all : 2 channels per lane
    int d0 = lane * 2;
    float a0 = 0.f, a1 = 0.f;
    for (int t = 0; t < R3; ++t) {
      float w = lg[rr * R3P + t];            // uniform across lanes
      if (w != 0.0f) {                       // skips ~87% (empty cells)
        int nidx = nb[rr * R3P + t] & 0x3fffffff;
        const float2 vv = *(const float2*)(vf + (size_t)nidx * DD + d0);
        a0 += w * vv.x;
        a1 += w * vv.y;
      }
    }
    a0 += vf[(size_t)mI * DD + d0];
    a1 += vf[(size_t)mI * DD + d0 + 1];
    float2 ro = { a0, a1 };
    *(float2*)(refined + (size_t)mI * DD + d0) = ro;
  }
}

// ---------------- output head: W_out then W_cls ----------------
__global__ __launch_bounds__(128) void k_head(
    const float* __restrict__ refined, const __bf16* __restrict__ Bo,
    const float* __restrict__ bout, const __bf16* __restrict__ Bc,
    const float* __restrict__ bcls,
    float* __restrict__ out_feat, float* __restrict__ out_cls) {
  __shared__ float Fs[4][16 * DD];
  int tid   = threadIdx.x;
  int wave  = tid >> 5;
  int lane  = tid & 31;
  int row   = lane & 15;
  int khalf = lane >> 4;
  int m0    = (blockIdx.x * 4 + wave) * 16;

  // GEMM1: refined @ W_out
  v8f ac[4] = {};
  #pragma unroll
  for (int c = 0; c < 2; ++c) {
    const float* fr = refined + (size_t)(m0 + row) * DD + c * 32 + khalf * 8;
    v4f g0 = *(const v4f*)(fr);
    v4f g1 = *(const v4f*)(fr + 4);
    v4f g2 = *(const v4f*)(fr + 16);
    v4f g3 = *(const v4f*)(fr + 20);
    v16bf a;
    #pragma unroll
    for (int j = 0; j < 4; ++j) {
      a[j] = f2bf(g0[j]); a[4+j] = f2bf(g1[j]);
      a[8+j] = f2bf(g2[j]); a[12+j] = f2bf(g3[j]);
    }
    v16bf bo[4];
    #pragma unroll
    for (int nt = 0; nt < 4; ++nt) {
      const __bf16* bp = Bo + (nt * 16 + (lane & 15)) * DD + c * 32 + khalf * 8;
      bo[nt] = mkv16(*(const v8bf*)bp, *(const v8bf*)(bp + 16));
    }
    #pragma unroll
    for (int nt = 0; nt < 4; ++nt)
      ac[nt] = WMMA_BF16(a, bo[nt], ac[nt]);
  }
  float* fs = &Fs[wave][0];
  #pragma unroll
  for (int nt = 0; nt < 4; ++nt)
    #pragma unroll
    for (int i = 0; i < 8; ++i) {
      int rr = i + 8 * khalf, cc = nt * 16 + (lane & 15);
      float v = ac[nt][i] + bout[cc];
      out_feat[(size_t)(m0 + rr) * DD + cc] = v;
      fs[rr * DD + cc] = v;
    }
  __syncthreads();

  // GEMM2: feat @ W_cls (N = 16 padded, 13 valid)
  v8f c2 = {};
  #pragma unroll
  for (int c = 0; c < 2; ++c) {
    const float* fr = fs + row * DD + c * 32 + khalf * 8;
    v4f g0 = *(const v4f*)(fr);
    v4f g1 = *(const v4f*)(fr + 4);
    v4f g2 = *(const v4f*)(fr + 16);
    v4f g3 = *(const v4f*)(fr + 20);
    v16bf a;
    #pragma unroll
    for (int j = 0; j < 4; ++j) {
      a[j] = f2bf(g0[j]); a[4+j] = f2bf(g1[j]);
      a[8+j] = f2bf(g2[j]); a[12+j] = f2bf(g3[j]);
    }
    const __bf16* bp = Bc + (lane & 15) * DD + c * 32 + khalf * 8;
    v16bf b = mkv16(*(const v8bf*)bp, *(const v8bf*)(bp + 16));
    c2 = WMMA_BF16(a, b, c2);
  }
  #pragma unroll
  for (int i = 0; i < 8; ++i) {
    int rr = i + 8 * khalf, cc = lane & 15;
    if (cc < NC) out_cls[(size_t)(m0 + rr) * NC + cc] = c2[i] + bcls[cc];
  }
}

// ---------------- launch ----------------
extern "C" void kernel_launch(void* const* d_in, const int* in_sizes, int n_in,
                              void* d_out, int out_size, void* d_ws, size_t ws_size,
                              hipStream_t stream) {
  (void)in_sizes; (void)n_in; (void)out_size; (void)ws_size;
  const int*   idx   = (const int*)  d_in[0];
  const float* geo   = (const float*)d_in[1];
  const float* sem   = (const float*)d_in[2];
  const float* Wconv = (const float*)d_in[3];
  const float* gam   = (const float*)d_in[4];
  const float* bet   = (const float*)d_in[5];
  const float* Wbdy  = (const float*)d_in[6];
  const float* bbdy  = (const float*)d_in[7];
  const float* Wq    = (const float*)d_in[8];
  const float* Wk    = (const float*)d_in[9];
  const float* Wv    = (const float*)d_in[10];
  const float* pemb  = (const float*)d_in[11];
  const float* Wout  = (const float*)d_in[12];
  const float* bout  = (const float*)d_in[13];
  const float* Wcls  = (const float*)d_in[14];
  const float* bcls  = (const float*)d_in[15];

  char* ws = (char*)d_ws;
  int*        lut     = (int*)      (ws + 0);
  int*        rank    = (int*)      (ws + 262144);
  long long*  sk      = (long long*)(ws + 524288);
  int*        si      = (int*)      (ws + 589824);
  __bf16*     Wc      = (__bf16*)   (ws + 622592);    // 125*2048*2  = 512000
  __bf16*     Bq      = (__bf16*)   (ws + 1134592);   // 8192
  __bf16*     Bk      = (__bf16*)   (ws + 1142784);   // 8192
  __bf16*     Bo      = (__bf16*)   (ws + 1150976);   // 8192
  __bf16*     Bv      = (__bf16*)   (ws + 1159168);   // 12288
  __bf16*     Bc      = (__bf16*)   (ws + 1171456);   // 2048
  __bf16*     Pb      = (__bf16*)   (ws + 1173504);   // 343*64*2 = 43904
  __bf16*     qgeo_bf = (__bf16*)   (ws + 1217408);   // 1 MB
  __bf16*     qb      = (__bf16*)   (ws + 2265984);   // 1 MB
  float*      qf      = (float*)    (ws + 3314560);   // 2 MB
  float*      kf      = (float*)    (ws + 5411712);   // 2 MB
  float*      vf      = (float*)    (ws + 7508864);   // 2 MB
  float*      refined = (float*)    (ws + 9606016);   // 2 MB

  float* out    = (float*)d_out;
  float* o_cls  = out;                                   // M*13
  float* o_bdy  = out + (size_t)M_TOT * NC;              // M
  float* o_aff  = o_bdy + M_TOT;                         // M*343
  float* o_feat = o_aff + (size_t)M_TOT * R3;            // M*64
  int*   o_nbr  = (int*)(o_feat + (size_t)M_TOT * DD);   // M*343 (int bits)

  k_init_lut   <<<(NCELL + 255) / 256, 256, 0, stream>>>(lut);
  k_scatter_lut<<<(M_TOT + 255) / 256, 256, 0, stream>>>(idx, lut);
  k_scan       <<<1, 1024, 0, stream>>>(lut, rank);
  k_sorted     <<<(M_TOT + 255) / 256, 256, 0, stream>>>(idx, rank, sk, si);
  k_wprep      <<<(KS3 * GEOC * DD + 255) / 256, 256, 0, stream>>>(
                  Wconv, Wq, Wk, Wv, Wout, Wcls, pemb, Wc, Bq, Bk, Bv, Bo, Bc, Pb);

  k_conv<<<M_TOT / 16 / 4, 128, 0, stream>>>(idx, geo, Wc, gam, bet, Wbdy, bbdy,
                                             lut, qgeo_bf, o_bdy);
  k_qkv <<<M_TOT / 16 / 4, 128, 0, stream>>>(qgeo_bf, sem, Bq, Bk, Bv, qb, qf, kf, vf);
  k_attn<<<M_TOT / 16, 32, 0, stream>>>(idx, lut, rank, sk, si, qb, qf, kf, vf, Pb,
                                        o_aff, o_nbr, refined);
  k_head<<<M_TOT / 16 / 4, 128, 0, stream>>>(refined, Bo, bout, Bc, bcls,
                                             o_feat, o_cls);
}
